// KNNTransformer_24240795418774
// MI455X (gfx1250) — matrix operand order repaired and checked by reading, not compiled, not run on hardware
//
#include <hip/hip_runtime.h>
#include <hip/hip_bf16.h>

// Problem constants (from reference): B=8, Q=128, D=2048, N=5120, k=1024
static constexpr int cB = 8;
static constexpr int cQ = 128;
static constexpr int cD = 2048;
static constexpr int cN = 5120;
static constexpr int cK = 1024;     // top-k
static constexpr int cWIN = 2;
static constexpr int cMAXLEN = 512;
static constexpr int PAD_ID = -1;

typedef __attribute__((ext_vector_type(16))) __bf16 v16bf;
typedef __attribute__((ext_vector_type(8)))  float  v8f;

typedef __attribute__((ext_vector_type(4))) unsigned int tdm_u32x4;
typedef __attribute__((ext_vector_type(8))) int          tdm_i32x8;
typedef __attribute__((ext_vector_type(4))) int          tdm_i32x4;

#if defined(__has_builtin)
#if __has_builtin(__builtin_amdgcn_tensor_load_to_lds) && \
    __has_builtin(__builtin_amdgcn_s_wait_tensorcnt)
#define HAVE_TDM 1
#endif
#endif
#ifndef HAVE_TDM
#define HAVE_TDM 0
#endif

// Native converts: gfx1250 has BF16 VALU support, let the backend pick
// v_cvt_*bf16* instead of a manual bfe/add3 RNE sequence.
__device__ __forceinline__ __bf16 f2bf(float f) { return (__bf16)f; }

union Frag32B {                 // assemble a v16bf from two 16B LDS loads
    uint4 q[2];
    v16bf v;
};

#if HAVE_TDM
// TDM 2D tile load: 128 rows x 64 bf16 (16 KB), row stride cD elements,
// from global `gsrc` (tile start) into LDS byte offset `lds_off`.
__device__ __forceinline__ void tdm_load_a(const __bf16* gsrc, unsigned lds_off) {
    const unsigned long long ga = (unsigned long long)(uintptr_t)gsrc;
    tdm_u32x4 g0 = {
        1u,                                                  // count=1 (valid D#)
        lds_off,                                             // lds_addr
        (unsigned)(ga & 0xFFFFFFFFull),                      // global_addr[31:0]
        (unsigned)((ga >> 32) & 0x1FFFFFFull) | (2u << 30)   // addr[56:32] | type=2
    };
    tdm_i32x8 g1 = {
        (int)0x00010000u,          // workgroup_mask=0, data_size=1 (2 bytes)
        (int)((unsigned)cD << 16), // tensor_dim0[15:0]  @ bits 63:48
        (int)(128u << 16),         // tensor_dim1[15:0]  @ bits 95:80
        (int)(64u  << 16),         // tile_dim0 = 64     @ bits 127:112
        128,                       // tile_dim1 = 128    @ bits 143:128
        cD,                        // tensor_dim0_stride @ bits 207:160 (lo32)
        0, 0
    };
#if __clang_major__ >= 23
    __builtin_amdgcn_tensor_load_to_lds(g0, g1, (tdm_i32x4)0, (tdm_i32x4)0,
                                        (tdm_i32x8)0, 0);
#else
    __builtin_amdgcn_tensor_load_to_lds(g0, g1, (tdm_i32x4)0, (tdm_i32x4)0, 0);
#endif
}
#endif

// ---------------------------------------------------------------------------
// Kernel 1: queries fp32 -> bf16 + ||q||^2 per row. grid = B*Q x 256.
// ---------------------------------------------------------------------------
__launch_bounds__(256)
__global__ void prep_q_kernel(const float* __restrict__ qin,
                              __bf16* __restrict__ qbf,
                              float* __restrict__ qsq) {
    const int row = blockIdx.x;                       // b*Q + q
    const float* src = qin + (size_t)row * cD;
    __bf16* dst = qbf + (size_t)row * cD;
    float s = 0.f;
    for (int i = threadIdx.x; i < cD; i += blockDim.x) {
        float v = src[i];
        dst[i] = f2bf(v);
        s += v * v;
    }
    __shared__ float red[256];
    red[threadIdx.x] = s;
    __syncthreads();
    for (int off = 128; off > 0; off >>= 1) {
        if ((int)threadIdx.x < off) red[threadIdx.x] += red[threadIdx.x + off];
        __syncthreads();
    }
    if (threadIdx.x == 0) qsq[row] = red[0];
}

// ---------------------------------------------------------------------------
// Kernel 2: batched GEMM + distance epilogue, v_wmma_f32_16x16x32_bf16.
// Block tile 128(Q) x 64(N), K-step 64 => 8 WMMAs per barrier pair.
// A tiles (128x64 bf16, 16 KB) arrive via double-buffered TDM; the next tile's
// DMA stays in flight during compute (s_wait_tensorcnt 1). B tile (64x64 fp32)
// is streamed from HBM once, converted with native bf16 casts, and stored to
// LDS in WMMA fragment layout via ds_store_b64 so lanes read ds_load_b128.
// ---------------------------------------------------------------------------
__launch_bounds__(256)
__global__ void gemm_dist_kernel(const __bf16* __restrict__ qbf,   // [B,Q,D]
                                 const float*  __restrict__ tok,    // [B,D,N]
                                 const float*  __restrict__ tsq,    // [B,N]
                                 const int*    __restrict__ tmask,  // [B,N]
                                 const float*  __restrict__ qsq,    // [B,Q]
                                 float* __restrict__ dist)          // [B,Q,N]
{
    // As is FIRST in LDS => byte offset 0 / 16384 (TDM D#.lds_addr).
    __shared__ __align__(16) __bf16 As[2][128 * 64];   // 2 x 16 KB row-major tiles
    __shared__ __align__(16) __bf16 Bt[2][4][32][16];  // 8 KB: [kk][sub][lane][elem]

    const int b   = blockIdx.y;
    const int n0  = blockIdx.x * 64;
    const int tid = threadIdx.x;
    const int lane  = tid & 31;
    const int wave  = tid >> 5;            // 0..7 -> q subtile
    const int mcol  = lane & 15;
    const int khalf = lane >> 4;           // which K-half this lane holds

    v8f acc[4] = {v8f{0.f,0.f,0.f,0.f,0.f,0.f,0.f,0.f},
                  v8f{0.f,0.f,0.f,0.f,0.f,0.f,0.f,0.f},
                  v8f{0.f,0.f,0.f,0.f,0.f,0.f,0.f,0.f},
                  v8f{0.f,0.f,0.f,0.f,0.f,0.f,0.f,0.f}};

    // B staging: thread owns a 4(k) x 4(n) micro-tile of the 64x64 B tile.
    const int kp = tid >> 4;               // 0..15 -> k rows 4kp..4kp+3
    const int nc = (tid & 15) * 4;         // n col base 0..60
    const int krem = (4 * kp) & 31;
    const int bkk  = kp >> 3;              // which 32-wide K half of the tile
    const int lgrp = (krem >> 4) << 4;     // fragment lane group
    const int eb   = krem & 15;            // fragment elem base (multiple of 4)

    const __bf16* a_base = qbf + ((size_t)(b * cQ)) * cD;   // batch A block

#if HAVE_TDM
    if (wave == 0) tdm_load_a(a_base, 0u);                  // prologue: buf 0
#endif

    for (int k0 = 0; k0 < cD; k0 += 64) {
        const int buf = (k0 >> 6) & 1;
        __syncthreads();   // previous iteration's LDS reads complete

#if HAVE_TDM
        if (wave == 0 && (k0 + 64) < cD)
            tdm_load_a(a_base + (k0 + 64), (unsigned)((buf ^ 1) * 16384));
#else
        {   // fallback: cooperative vector load of the A tile (128x64)
            const int ar = tid >> 1;
            const int ac = (tid & 1) * 32;
            const uint4* s4 = (const uint4*)(a_base + (size_t)ar * cD + k0 + ac);
            uint4* d4 = (uint4*)&As[buf][ar * 64 + ac];
            d4[0] = s4[0]; d4[1] = s4[1]; d4[2] = s4[2]; d4[3] = s4[3];
        }
#endif
        // --- stage B tile fp32 -> bf16 into fragment layout ---
        {
            const float* srow = tok + ((size_t)b * cD + (k0 + 4 * kp)) * cN + n0 + nc;
            float4 r0 = *(const float4*)(srow);
            float4 r1 = *(const float4*)(srow + cN);
            float4 r2 = *(const float4*)(srow + 2 * (size_t)cN);
            float4 r3 = *(const float4*)(srow + 3 * (size_t)cN);
            float c0[4] = {r0.x, r0.y, r0.z, r0.w};
            float c1[4] = {r1.x, r1.y, r1.z, r1.w};
            float c2[4] = {r2.x, r2.y, r2.z, r2.w};
            float c3[4] = {r3.x, r3.y, r3.z, r3.w};
            #pragma unroll
            for (int j = 0; j < 4; ++j) {
                const int nl = nc + j;
                union { __bf16 h[4]; unsigned long long u; } pk;
                pk.h[0] = f2bf(c0[j]); pk.h[1] = f2bf(c1[j]);
                pk.h[2] = f2bf(c2[j]); pk.h[3] = f2bf(c3[j]);
                // 4 contiguous K elems of one fragment lane: single b64 store
                *(unsigned long long*)&Bt[bkk][nl >> 4][lgrp | (nl & 15)][eb] = pk.u;
            }
        }

        // prefetch next tok tile (global_prefetch_b8)
        if (k0 + 64 < cD) {
            const float* nsrc = tok + ((size_t)b * cD + (k0 + 64 + 4 * kp)) * cN + n0 + nc;
            __builtin_prefetch(nsrc, 0, 0);
        }

#if HAVE_TDM
        if (wave == 0) {
            if ((k0 + 64) < cD) __builtin_amdgcn_s_wait_tensorcnt(1); // next in flight
            else                __builtin_amdgcn_s_wait_tensorcnt(0);
        }
#endif
        __syncthreads();   // As[buf] (TDM) + Bt visible to all waves

        #pragma unroll
        for (int kk = 0; kk < 2; ++kk) {
            // A fragment: two contiguous 16B chunks per lane
            Frag32B af;
            const __bf16* arow = &As[buf][(wave * 16 + mcol) * 64 + kk * 32 + khalf * 16];
            af.q[0] = *(const uint4*)(arow);        // K kb..kb+7
            af.q[1] = *(const uint4*)(arow + 8);    // K kb+8..kb+15
            // Load all 4 B fragments first, then issue 4 back-to-back WMMAs
            Frag32B bf0, bf1, bf2, bf3;
            bf0.q[0] = *(const uint4*)&Bt[kk][0][lane][0];
            bf0.q[1] = *(const uint4*)&Bt[kk][0][lane][8];
            bf1.q[0] = *(const uint4*)&Bt[kk][1][lane][0];
            bf1.q[1] = *(const uint4*)&Bt[kk][1][lane][8];
            bf2.q[0] = *(const uint4*)&Bt[kk][2][lane][0];
            bf2.q[1] = *(const uint4*)&Bt[kk][2][lane][8];
            bf3.q[0] = *(const uint4*)&Bt[kk][3][lane][0];
            bf3.q[1] = *(const uint4*)&Bt[kk][3][lane][8];
            acc[0] = __builtin_amdgcn_wmma_f32_16x16x32_bf16(
                false, af.v, false, bf0.v, (short)0, acc[0], false, false);
            acc[1] = __builtin_amdgcn_wmma_f32_16x16x32_bf16(
                false, af.v, false, bf1.v, (short)0, acc[1], false, false);
            acc[2] = __builtin_amdgcn_wmma_f32_16x16x32_bf16(
                false, af.v, false, bf2.v, (short)0, acc[2], false, false);
            acc[3] = __builtin_amdgcn_wmma_f32_16x16x32_bf16(
                false, af.v, false, bf3.v, (short)0, acc[3], false, false);
        }
    }

    // --- epilogue: dist = tsq[n] + qsq[q] - 2*acc, masked -> +inf ---
    float qv[8];
    #pragma unroll
    for (int r = 0; r < 8; ++r)
        qv[r] = qsq[b * cQ + wave * 16 + khalf * 8 + r];

    #pragma unroll
    for (int sub = 0; sub < 4; ++sub) {
        const int n = n0 + sub * 16 + mcol;
        const float ts = tsq[(size_t)b * cN + n];
        const int  mk  = tmask[(size_t)b * cN + n];
        #pragma unroll
        for (int r = 0; r < 8; ++r) {
            const int q = wave * 16 + khalf * 8 + r;   // C/D layout: M = r + 8*khalf
            float dv = ts + qv[r] - 2.0f * acc[sub][r];
            if (mk <= 0) dv = __builtin_inff();
            dist[((size_t)(b * cQ + q)) * cN + n] = dv;
        }
    }
}

// ---------------------------------------------------------------------------
// Kernel 3: row-wise top-k (1024 of 5120): bitonic sort of 8192 packed
// (monotone-float<<32 | idx) u64 keys in 64 KB LDS. grid = B*Q x 1024.
// ---------------------------------------------------------------------------
__launch_bounds__(1024)
__global__ void topk_kernel(const float* __restrict__ dist,
                            float* __restrict__ neg_out,   // [B*Q, k]
                            int*   __restrict__ idx_out)   // [B*Q, k]
{
    __shared__ unsigned long long keys[8192];
    const int row = blockIdx.x;
    const float* src = dist + (size_t)row * cN;

    for (int i = threadIdx.x; i < 8192; i += blockDim.x) {
        unsigned u;
        if (i < cN) { union { float f; unsigned u; } c; c.f = src[i]; u = c.u; }
        else        { union { float f; unsigned u; } c; c.f = __builtin_inff(); u = c.u; }
        unsigned key = (u & 0x80000000u) ? ~u : (u | 0x80000000u);
        keys[i] = ((unsigned long long)key << 32) | (unsigned)i;
    }
    __syncthreads();

    for (int size = 2; size <= 8192; size <<= 1) {
        for (int stride = size >> 1; stride > 0; stride >>= 1) {
            for (int i = threadIdx.x; i < 8192; i += blockDim.x) {
                const int j = i ^ stride;
                if (j > i) {
                    const bool up = (i & size) == 0;
                    unsigned long long a = keys[i], c = keys[j];
                    if ((a > c) == up) { keys[i] = c; keys[j] = a; }
                }
            }
            __syncthreads();
        }
    }

    for (int t = threadIdx.x; t < cK; t += blockDim.x) {
        const unsigned long long kv = keys[t];
        const unsigned key = (unsigned)(kv >> 32);
        const unsigned u = (key & 0x80000000u) ? (key & 0x7FFFFFFFu) : ~key;
        union { unsigned u; float f; } c; c.u = u;
        neg_out[(size_t)row * cK + t] = -c.f;              // top_k(-dist) values
        idx_out[(size_t)row * cK + t] = (int)(kv & 0xFFFFFFFFu);
    }
}

// ---------------------------------------------------------------------------
// Kernel 4: window-2 n-gram gather + validity fix; emit as float.
// ---------------------------------------------------------------------------
__launch_bounds__(256)
__global__ void gather_kernel(const int* __restrict__ idx_in,   // [B*Q*k]
                              const int* __restrict__ vals,     // [B*N]
                              const int* __restrict__ knn,      // [B*N]
                              float* __restrict__ tok_out,      // [B*Q*k*2]
                              float* __restrict__ doc_out,
                              float* __restrict__ pos_out)
{
    const int g = blockIdx.x * blockDim.x + threadIdx.x;
    if (g >= cB * cQ * cK) return;
    const int row = g / cK;
    const int b = row / cQ;
    const int idx = idx_in[g];

    int tk[cWIN], dc[cWIN], ps[cWIN];
    #pragma unroll
    for (int w = 0; w < cWIN; ++w) {
        const int ii = idx + w;
        long long tv, kv;
        if (ii < cN) { tv = vals[(size_t)b * cN + ii]; kv = knn[(size_t)b * cN + ii]; }
        else         { tv = PAD_ID; kv = PAD_ID; }      // tail padding columns
        long long dq = (kv >= 0) ? (kv / cMAXLEN)
                                 : -(((-kv) + cMAXLEN - 1) / cMAXLEN);  // floor div
        long long dr = kv - dq * cMAXLEN;
        tk[w] = (int)tv; dc[w] = (int)dq; ps[w] = (int)dr;
    }
    if (!(ps[1] > ps[0])) { tk[1] = PAD_ID; dc[1] = PAD_ID; ps[1] = PAD_ID; }

    const size_t base = (size_t)g * cWIN;
    tok_out[base] = (float)tk[0]; tok_out[base + 1] = (float)tk[1];
    doc_out[base] = (float)dc[0]; doc_out[base + 1] = (float)dc[1];
    pos_out[base] = (float)ps[0]; pos_out[base + 1] = (float)ps[1];
}

// ---------------------------------------------------------------------------
extern "C" void kernel_launch(void* const* d_in, const int* in_sizes, int n_in,
                              void* d_out, int out_size, void* d_ws, size_t ws_size,
                              hipStream_t stream) {
    (void)in_sizes; (void)n_in; (void)out_size; (void)ws_size;

    const float* q     = (const float*)d_in[0];   // [B,Q,D]
    const float* tok   = (const float*)d_in[1];   // [B,D,N]
    const float* tsq   = (const float*)d_in[2];   // [B,1,N]
    const int*   tmask = (const int*)  d_in[3];   // [B,1,N]
    const int*   vals  = (const int*)  d_in[4];   // [B,1,N]
    const int*   knn   = (const int*)  d_in[5];   // [B,1,N]
    float* out = (float*)d_out;

    char* ws = (char*)d_ws;
    size_t off = 0;
    __bf16* qbf = (__bf16*)(ws + off); off += (size_t)cB * cQ * cD * sizeof(__bf16);
    float*  qsq = (float*)(ws + off);  off += (size_t)cB * cQ * sizeof(float);
    off = (off + 255) & ~(size_t)255;
    float*  dist = (float*)(ws + off); off += (size_t)cB * cQ * cN * sizeof(float);
    int*    idxb = (int*)(ws + off);

    prep_q_kernel<<<cB * cQ, 256, 0, stream>>>(q, qbf, qsq);

    dim3 ggrid(cN / 64, cB);
    gemm_dist_kernel<<<ggrid, 256, 0, stream>>>(qbf, tok, tsq, tmask, qsq, dist);

    topk_kernel<<<cB * cQ, 1024, 0, stream>>>(dist, out, idxb);

    float* tok_out = out + (size_t)cB * cQ * cK;
    float* doc_out = tok_out + (size_t)cB * cQ * cK * cWIN;
    float* pos_out = doc_out + (size_t)cB * cQ * cK * cWIN;
    const int total = cB * cQ * cK;
    gather_kernel<<<(total + 255) / 256, 256, 0, stream>>>(idxb, vals, knn,
                                                           tok_out, doc_out, pos_out);
}
